// BinASPP_22273700397584
// MI455X (gfx1250) — compile-verified, matrix operand order checked by reading, not run on hardware
//
#include <hip/hip_runtime.h>
#include <hip/hip_bf16.h>

// ---------------------------------------------------------------------------
// BinASPP on MI455X (gfx1250): binarized convs == signed-int8 (+1/0/-1) GEMMs
// -> V_WMMA_I32_16X16X64_IU8, with the shared B panel staged in LDS through
// the CDNA5 async global->LDS path (ASYNCcnt), double-buffered per tap.
// ---------------------------------------------------------------------------

typedef int   v8i __attribute__((ext_vector_type(8)));
typedef short v8s __attribute__((ext_vector_type(8)));

#define EPSV 1e-5f

__device__ __forceinline__ signed char sgn8(float v) {
    return (v > 0.f) ? (signed char)1 : ((v < 0.f) ? (signed char)-1 : (signed char)0);
}
__device__ __forceinline__ float fsgn(float v) {
    return (v > 0.f) ? 1.f : ((v < 0.f) ? -1.f : 0.f);
}

// Problem sizes
constexpr int N_ = 8, C_ = 256, H_ = 64, W_ = 64;
constexpr int SP_ = H_ * W_;          // 4096
constexpr int M_  = N_ * SP_;         // 32768 output pixels (GEMM M)

// Workspace layout (bytes). Total ~94.8 MB.
constexpr size_t OFF_XQ    = 0;                              // int8 [M][C]
constexpr size_t SZ_XQ     = (size_t)M_ * C_;                // 8,388,608
constexpr size_t OFF_W1Q   = OFF_XQ + SZ_XQ;                 // int8 [C][C]
constexpr size_t SZ_W1Q    = (size_t)C_ * C_;
constexpr size_t OFF_W3Q   = OFF_W1Q + SZ_W1Q;               // int8 [4][9][C][C]
constexpr size_t SZ_W3Q    = (size_t)4 * 9 * C_ * C_;
constexpr size_t OFF_Y     = OFF_W3Q + SZ_W3Q;               // int16 [5][C][M]
constexpr size_t SZ_Y      = (size_t)5 * C_ * M_ * 2;
constexpr size_t OFF_MEANX = OFF_Y + SZ_Y;                   // f32 [N][C]
constexpr size_t SZ_MEANX  = (size_t)N_ * C_ * 4;
constexpr size_t OFF_SPOOL = OFF_MEANX + SZ_MEANX;           // f32 [N][C]
constexpr size_t SZ_SPOOL  = (size_t)N_ * C_ * 4;
constexpr size_t OFF_SUMS  = OFF_SPOOL + SZ_SPOOL;           // i32 [5][C]
constexpr size_t SZ_SUMS   = (size_t)5 * C_ * 4;
constexpr size_t OFF_SUMSQ = OFF_SUMS + SZ_SUMS;             // f64 [5][C] (8B aligned)
constexpr size_t SZ_SUMSQ  = (size_t)5 * C_ * 8;
constexpr size_t OFF_AC    = OFF_SUMSQ + SZ_SUMSQ;           // f32 [5][C]
constexpr size_t SZ_AC     = (size_t)5 * C_ * 4;
constexpr size_t OFF_BC    = OFF_AC + SZ_AC;                 // f32 [5][C]

// Load an 8-VGPR int8 fragment (ISA 8-bit 16x64 layout: per lane, K bytes at
// base + half*8 in 8B chunks spaced 16B apart). Works for global or LDS ptrs.
#define LOAD_FRAG8(dst, ptr)                                                 \
    do {                                                                     \
        int2 _d0 = *reinterpret_cast<const int2*>((ptr));                    \
        int2 _d1 = *reinterpret_cast<const int2*>((ptr) + 16);               \
        int2 _d2 = *reinterpret_cast<const int2*>((ptr) + 32);               \
        int2 _d3 = *reinterpret_cast<const int2*>((ptr) + 48);               \
        (dst)[0] = _d0.x; (dst)[1] = _d0.y;                                  \
        (dst)[2] = _d1.x; (dst)[3] = _d1.y;                                  \
        (dst)[4] = _d2.x; (dst)[5] = _d2.y;                                  \
        (dst)[6] = _d3.x; (dst)[7] = _d3.y;                                  \
    } while (0)

// ---------------------------------------------------------------------------
// Quantization kernels
// ---------------------------------------------------------------------------

__global__ __launch_bounds__(256) void k_quant_x(const float* __restrict__ x,
                                                 signed char* __restrict__ xq) {
    int idx = blockIdx.x * 256 + threadIdx.x;          // over M_*C_
    int n  = idx >> 20;
    int sp = (idx >> 8) & 4095;
    int c  = idx & 255;
    xq[idx] = sgn8(x[((size_t)n << 20) + ((size_t)c << 12) + sp]);
}

__global__ __launch_bounds__(256) void k_quant_w1(const float* __restrict__ w1,
                                                  signed char* __restrict__ w1q) {
    int idx = blockIdx.x * 256 + threadIdx.x;          // over C_*C_
    w1q[idx] = sgn8(w1[idx]);                          // [Cout][Cin] (1x1)
}

// w3: [4][Cout][Cin][3][3] -> w3q: [br][tap][Cout][Cin]
__global__ __launch_bounds__(256) void k_quant_w3(const float* __restrict__ w3,
                                                  signed char* __restrict__ w3q) {
    int idx = blockIdx.x * 256 + threadIdx.x;
    if (idx >= 4 * 9 * C_ * C_) return;
    int br  = idx / (9 * C_ * C_);
    int rem = idx % (9 * C_ * C_);
    int tap = rem / (C_ * C_);
    int r2  = rem % (C_ * C_);
    int co  = r2 >> 8;
    int ci  = r2 & 255;
    size_t src = (((size_t)(br * C_ + co) * C_) + ci) * 9 + tap;
    w3q[idx] = sgn8(w3[src]);
}

// ---------------------------------------------------------------------------
// Pooling branch
// ---------------------------------------------------------------------------

__global__ __launch_bounds__(256) void k_meanx(const float* __restrict__ x,
                                               float* __restrict__ meanx) {
    __shared__ float red[256];
    int b = blockIdx.x;                                // n*256 + c
    const float* p = x + (size_t)b * SP_;
    float s = 0.f;
    for (int i = threadIdx.x; i < SP_; i += 256) s += p[i];
    red[threadIdx.x] = s;
    __syncthreads();
    for (int off = 128; off > 0; off >>= 1) {
        if ((int)threadIdx.x < off) red[threadIdx.x] += red[threadIdx.x + off];
        __syncthreads();
    }
    if (threadIdx.x == 0) meanx[b] = red[0] / (float)SP_;
}

__global__ __launch_bounds__(256) void k_pool(const float* __restrict__ meanx,
                                              const float* __restrict__ wpool,
                                              const float* __restrict__ gp,
                                              const float* __restrict__ bp,
                                              float* __restrict__ spool) {
    int co = threadIdx.x;
    float p[N_];
    for (int n = 0; n < N_; ++n) {
        float acc = 0.f;
        for (int ci = 0; ci < C_; ++ci)
            acc += fsgn(meanx[n * C_ + ci]) * fsgn(wpool[co * C_ + ci]);
        p[n] = acc;
    }
    float mu = 0.f;
    for (int n = 0; n < N_; ++n) mu += p[n];
    mu *= (1.f / N_);
    float var = 0.f;
    for (int n = 0; n < N_; ++n) { float d = p[n] - mu; var += d * d; }
    var *= (1.f / N_);
    float inv = rsqrtf(var + EPSV);
    for (int n = 0; n < N_; ++n)
        spool[n * C_ + co] = gp[co] * (p[n] - mu) * inv + bp[co];
}

__global__ void k_zero_stats(int* __restrict__ sumS, double* __restrict__ sumSq) {
    int idx = blockIdx.x * 256 + threadIdx.x;          // over 5*256
    if (idx < 5 * C_) { sumS[idx] = 0; sumSq[idx] = 0.0; }
}

// ---------------------------------------------------------------------------
// Implicit-GEMM int8 WMMA kernel.
// Block = 256 threads = 8 waves; block tile M=128 x N=64; wave: M=16 x N=64
// (4 independent accumulators -> 4 back-to-back v_wmma per K step).
// Per-tap 16KB B panel (shared by all 8 waves) is streamed into LDS with
// global_load_async_to_lds_b128 (ASYNCcnt), double-buffered so tap t+1
// prefetches while tap t computes; B fragments then come from ds_load_b64.
// ---------------------------------------------------------------------------
template <int NTAPS>
__global__ __launch_bounds__(256) void k_gemm(const signed char* __restrict__ xq,
                                              const signed char* __restrict__ wq,  // [NTAPS][256][256]
                                              short* __restrict__ ybr,             // [256][32768]
                                              int* __restrict__ sumS,
                                              double* __restrict__ sumSq,
                                              int rate) {
    __shared__ signed char bsh[2][64 * 256];   // 2 x 16KB B panels

    const int tid  = threadIdx.x;
    const int lane = tid & 31;
    const int wave = tid >> 5;
    const int row  = lane & 15;
    const int half = lane >> 4;
    const int n0   = blockIdx.x * 64;
    const int m0   = blockIdx.y * 128 + wave * 16;
    const int m    = m0 + row;
    const int nimg = m >> 12;
    const int sp   = m & 4095;
    const int yp   = sp >> 6;
    const int xp   = sp & 63;

    // Async prefetch of tap 0 B panel: 16KB = 256 threads x 4 x b128.
    {
        const signed char* src = wq + ((size_t)n0 << 8) + tid * 64;
        unsigned dst = (unsigned)(uintptr_t)(&bsh[0][tid * 64]);
#pragma unroll
        for (int j = 0; j < 4; ++j)
            asm volatile("global_load_async_to_lds_b128 %0, %1, off"
                         :: "v"(dst + j * 16), "v"(src + j * 16) : "memory");
    }
    asm volatile("s_wait_asynccnt 0" ::: "memory");
    __syncthreads();

    v8i acc0, acc1, acc2, acc3;
#pragma unroll
    for (int v = 0; v < 8; ++v) { acc0[v] = 0; acc1[v] = 0; acc2[v] = 0; acc3[v] = 0; }

    int buf = 0;
    for (int tap = 0; tap < NTAPS; ++tap) {
        // Prefetch next tap's B panel into the other LDS buffer.
        if (tap + 1 < NTAPS) {
            const signed char* src =
                wq + (((size_t)(tap + 1) * C_ + n0) << 8) + tid * 64;
            unsigned dst = (unsigned)(uintptr_t)(&bsh[buf ^ 1][tid * 64]);
#pragma unroll
            for (int j = 0; j < 4; ++j)
                asm volatile("global_load_async_to_lds_b128 %0, %1, off"
                             :: "v"(dst + j * 16), "v"(src + j * 16) : "memory");
        }

        const int dy = (NTAPS == 1) ? 0 : ((tap / 3) - 1) * rate;
        const int dx = (NTAPS == 1) ? 0 : ((tap % 3) - 1) * rate;
        const int y2 = yp + dy;
        const int x2 = xp + dx;
        const bool inb = ((unsigned)y2 < (unsigned)H_) && ((unsigned)x2 < (unsigned)W_);
        const signed char* aBase =
            xq + (((size_t)nimg << 20) + ((size_t)((y2 << 6) + x2) << 8));

#pragma unroll
        for (int k0 = 0; k0 < 256; k0 += 64) {
            v8i a;
            if (inb) {
                LOAD_FRAG8(a, (const char*)aBase + k0 + half * 8);
            } else {
#pragma unroll
                for (int v = 0; v < 8; ++v) a[v] = 0;
            }
            const signed char* bb = &bsh[buf][(row << 8) + k0 + half * 8];
            v8i b0, b1, b2, b3;
            LOAD_FRAG8(b0, bb);
            LOAD_FRAG8(b1, bb + 16 * 256);
            LOAD_FRAG8(b2, bb + 32 * 256);
            LOAD_FRAG8(b3, bb + 48 * 256);
            // 4 independent WMMAs back-to-back (hazard-free w.r.t. each other)
            acc0 = __builtin_amdgcn_wmma_i32_16x16x64_iu8(true, a, true, b0, acc0, false, false);
            acc1 = __builtin_amdgcn_wmma_i32_16x16x64_iu8(true, a, true, b1, acc1, false, false);
            acc2 = __builtin_amdgcn_wmma_i32_16x16x64_iu8(true, a, true, b2, acc2, false, false);
            acc3 = __builtin_amdgcn_wmma_i32_16x16x64_iu8(true, a, true, b3, acc3, false, false);
        }

        // Wait for next panel + make sure all waves are done with this one.
        asm volatile("s_wait_asynccnt 0" ::: "memory");
        __syncthreads();
        buf ^= 1;
    }

    // D layout: lane holds column N=row, rows M = v + 8*half (contiguous in M).
    v8i accs[4] = {acc0, acc1, acc2, acc3};
#pragma unroll
    for (int s = 0; s < 4; ++s) {
        const int c = n0 + s * 16 + row;
        v8s outv;
        int psum = 0, psq = 0;
#pragma unroll
        for (int v = 0; v < 8; ++v) {
            int t = accs[s][v];
            outv[v] = (short)t;            // |y| <= 2304, fits int16
            psum += t;
            psq  += t * t;
        }
        *reinterpret_cast<v8s*>(ybr + (((size_t)c) << 15) + m0 + half * 8) = outv;
        psum += __shfl_xor(psum, 16, 32);
        psq  += __shfl_xor(psq, 16, 32);
        if (half == 0) {
            atomicAdd(&sumS[c], psum);
            atomicAdd(&sumSq[c], (double)psq);
        }
    }
}

// ---------------------------------------------------------------------------
// Fold BN stats into per-channel affine: y -> a*y + b
// ---------------------------------------------------------------------------
__global__ void k_bnprep(const int* __restrict__ sumS, const double* __restrict__ sumSq,
                         const float* __restrict__ g1, const float* __restrict__ b1,
                         const float* __restrict__ g3, const float* __restrict__ b3,
                         float* __restrict__ aC, float* __restrict__ bC) {
    int idx = blockIdx.x * 256 + threadIdx.x;          // over 5*256
    if (idx >= 5 * C_) return;
    int br = idx >> 8, c = idx & 255;
    float g, bb;
    if (br == 0) { g = g1[c]; bb = b1[c]; }
    else         { g = g3[(br - 1) * C_ + c]; bb = b3[(br - 1) * C_ + c]; }
    float mu  = (float)sumS[idx] / (float)M_;
    float var = (float)(sumSq[idx] / (double)M_) - mu * mu;
    float inv = rsqrtf(var + EPSV);
    aC[idx] = g * inv;
    bC[idx] = bb - g * mu * inv;
}

// ---------------------------------------------------------------------------
// Finalize: out[n,c,h,w] = spool[n,c] + sum_br htanh(a*y + b)
// ---------------------------------------------------------------------------
__global__ __launch_bounds__(256) void k_finalize(const short* __restrict__ ybuf,
                                                  const float* __restrict__ spool,
                                                  const float* __restrict__ aC,
                                                  const float* __restrict__ bC,
                                                  float* __restrict__ out) {
    int idx = blockIdx.x * 256 + threadIdx.x;          // over N*C*SP
    int n  = idx >> 20;
    int c  = (idx >> 12) & 255;
    int sp = idx & 4095;
    int m  = (n << 12) | sp;
    float s = spool[(n << 8) + c];
#pragma unroll
    for (int br = 0; br < 5; ++br) {
        float yv = (float)ybuf[(((size_t)(br * C_ + c)) << 15) + m];
        float t  = aC[br * C_ + c] * yv + bC[br * C_ + c];
        s += fminf(1.f, fmaxf(-1.f, t));
    }
    out[idx] = s;
}

// ---------------------------------------------------------------------------
extern "C" void kernel_launch(void* const* d_in, const int* in_sizes, int n_in,
                              void* d_out, int out_size, void* d_ws, size_t ws_size,
                              hipStream_t stream) {
    (void)in_sizes; (void)n_in; (void)out_size; (void)ws_size;
    const float* x     = (const float*)d_in[0];
    const float* wpool = (const float*)d_in[1];
    const float* gpool = (const float*)d_in[2];
    const float* bpool = (const float*)d_in[3];
    const float* w1    = (const float*)d_in[4];
    const float* g1    = (const float*)d_in[5];
    const float* b1    = (const float*)d_in[6];
    const float* w3    = (const float*)d_in[7];
    const float* g3    = (const float*)d_in[8];
    const float* b3    = (const float*)d_in[9];
    float* out = (float*)d_out;

    char* ws = (char*)d_ws;
    signed char* xq    = (signed char*)(ws + OFF_XQ);
    signed char* w1q   = (signed char*)(ws + OFF_W1Q);
    signed char* w3q   = (signed char*)(ws + OFF_W3Q);
    short*       ybuf  = (short*)(ws + OFF_Y);
    float*       meanx = (float*)(ws + OFF_MEANX);
    float*       spool = (float*)(ws + OFF_SPOOL);
    int*         sumS  = (int*)(ws + OFF_SUMS);
    double*      sumSq = (double*)(ws + OFF_SUMSQ);
    float*       aC    = (float*)(ws + OFF_AC);
    float*       bC    = (float*)(ws + OFF_BC);

    // Phase 0: quantize + pooling branch + stat init
    k_quant_x <<<(M_ * C_) / 256, 256, 0, stream>>>(x, xq);
    k_quant_w1<<<(C_ * C_) / 256, 256, 0, stream>>>(w1, w1q);
    k_quant_w3<<<(4 * 9 * C_ * C_ + 255) / 256, 256, 0, stream>>>(w3, w3q);
    k_meanx   <<<N_ * C_, 256, 0, stream>>>(x, meanx);
    k_pool    <<<1, 256, 0, stream>>>(meanx, wpool, gpool, bpool, spool);
    k_zero_stats<<<5, 256, 0, stream>>>(sumS, sumSq);

    // Phase 1: WMMA implicit GEMMs (branch 0 = 1x1; branches 1..4 = 3x3 dilated)
    dim3 gg(C_ / 64, M_ / 128);
    k_gemm<1><<<gg, 256, 0, stream>>>(xq, w1q, ybuf, sumS, sumSq, 1);
    const int rates[4] = {1, 4, 8, 12};
    for (int i = 0; i < 4; ++i) {
        int br = i + 1;
        k_gemm<9><<<gg, 256, 0, stream>>>(
            xq, w3q + (size_t)i * 9 * C_ * C_,
            ybuf + (size_t)br * C_ * M_,
            sumS + br * C_, sumSq + br * C_, rates[i]);
    }

    // Phase 2: BN fold + finalize
    k_bnprep  <<<5, 256, 0, stream>>>(sumS, sumSq, g1, b1, g3, b3, aC, bC);
    k_finalize<<<(M_ * C_) / 256, 256, 0, stream>>>(ybuf, spool, aC, bC, out);
}
